// DecoderRNN_22488448762372
// MI455X (gfx1250) — compile-verified, matrix operand order checked by reading, not run on hardware
//
#include <hip/hip_runtime.h>
#include <hip/hip_bf16.h>
#include <stdint.h>

// ---------------------------------------------------------------------------
// Show-Attend-Tell decoder for MI455X (gfx1250), wave32 + WMMA bf16.
// B=32, P=196, ENC=DEC=ATT=512, E=256, S=128, V=10000.
//
// All GEMMs are M=32 (two 16-row WMMA tiles) using v_wmma_f32_16x16x32_bf16
// with f32 accumulation. Weights are pre-packed once into the WMMA B-fragment
// per-lane layout so the GEMM inner loop is: 2x coalesced 32B loads + wmma.
// ---------------------------------------------------------------------------

typedef __attribute__((ext_vector_type(16))) __bf16 v16bf;
typedef __attribute__((ext_vector_type(8)))  float  v8f;

#define BB   32
#define PP   196
#define ENC  512
#define DEC  512
#define ATT  512
#define EE   256
#define SS   128
#define VV   10000
#define TSTEPS (SS - 1)          // 127

// K index held by (lane, i) inside a 16x32 (or 32x16) bf16 WMMA fragment.
// ISA 7.12.2: VGPR r holds K = 16*(i>=8) + 8*(lane>=16) + (i&7)  (i = 2r+parity)
__device__ __forceinline__ int kidx(int lane, int i) {
    return ((i >> 3) << 4) | (((lane >> 4) & 1) << 3) | (i & 7);
}

__device__ __forceinline__ float sigmoidf(float x) {
    return 1.0f / (1.0f + __expf(-x));
}

// ---------------------------------------------------------------------------
// Pack an f32 weight matrix into bf16 WMMA B-fragment layout.
//   B[k][n] = transposed ? src[n*lds + k] : src[k*lds + n]
// dst block (nb, kb+kbOffset): 32 lanes x 16 elems, lane-contiguous (32B/lane).
// One wave per (nb, kb) block.
// ---------------------------------------------------------------------------
__global__ void pack_B_kernel(const float* __restrict__ src, __bf16* __restrict__ dst,
                              int Ksrc, int N, int lds, int transposed,
                              int kbOffset, int KBtotal)
{
    int wave = blockIdx.x * (blockDim.x >> 5) + (threadIdx.x >> 5);
    int lane = threadIdx.x & 31;
    int KBsrc = Ksrc >> 5;
    int NB = N >> 4;
    if (wave >= NB * KBsrc) return;
    int nb = wave / KBsrc;
    int kb = wave % KBsrc;
    int n  = nb * 16 + (lane & 15);
    v16bf v;
    #pragma unroll
    for (int i = 0; i < 16; ++i) {
        int k = kb * 32 + kidx(lane, i);
        float x = transposed ? src[(int64_t)n * lds + k] : src[(int64_t)k * lds + n];
        v[i] = (__bf16)x;
    }
    *(v16bf*)(dst + ((((int64_t)nb * KBtotal) + (kb + kbOffset)) * 32 + lane) * 16) = v;
}

// Pack a [32,K] f32 activation matrix into bf16 WMMA A-fragment layout.
// dst index: ((kb*2 + mt)*32 + lane)*16 + i,   m = mt*16 + (lane&15).
__global__ void pack_A32_kernel(const float* __restrict__ src, __bf16* __restrict__ dst, int K)
{
    int idx = blockIdx.x * blockDim.x + threadIdx.x;       // over 32*K
    if (idx >= 32 * K) return;
    int m = idx / K, k = idx % K;
    int kb = k >> 5, kin = k & 31;
    int i    = ((kin >> 4) << 3) | (kin & 7);
    int lane = (m & 15) + (((kin >> 3) & 1) << 4);
    int mt   = m >> 4;
    dst[(((int64_t)kb * 2 + mt) * 32 + lane) * 16 + i] = (__bf16)src[idx];
}

// Pack features [32,196,512] into per-p A fragments: dst[p][kb][mt][lane][i].
__global__ void pack_feat_kernel(const float* __restrict__ f, __bf16* __restrict__ dst)
{
    int64_t idx = (int64_t)blockIdx.x * blockDim.x + threadIdx.x;  // b,p,k
    if (idx >= (int64_t)BB * PP * ENC) return;
    int k = (int)(idx % ENC);
    int64_t t = idx / ENC;
    int p = (int)(t % PP);
    int b = (int)(t / PP);
    int kb = k >> 5, kin = k & 31;
    int i    = ((kin >> 4) << 3) | (kin & 7);
    int lane = (b & 15) + (((kin >> 3) & 1) << 4);
    int mt   = b >> 4;
    dst[(int64_t)p * (16 * 1024) + (((int64_t)kb * 2 + mt) * 32 + lane) * 16 + i] = (__bf16)f[idx];
}

// mean over P of features -> meanf [32,512]
__global__ void mean_feat_kernel(const float* __restrict__ f, float* __restrict__ meanf)
{
    int idx = blockIdx.x * blockDim.x + threadIdx.x;       // 32*512
    if (idx >= BB * ENC) return;
    int b = idx >> 9, e = idx & 511;
    float s = 0.f;
    for (int p = 0; p < PP; ++p) s += f[((int64_t)b * PP + p) * ENC + e];
    meanf[idx] = s * (1.0f / PP);
}

__global__ void add_bias_kernel(const float* __restrict__ a, const float* __restrict__ b,
                                float* __restrict__ out, int n)
{
    int i = blockIdx.x * blockDim.x + threadIdx.x;
    if (i < n) out[i] = a[i] + b[i];
}

// ---------------------------------------------------------------------------
// Generic M=32 bf16 WMMA GEMM:  C[32,N] = concat_K(A0,A1) x B + bias
// A0: KB0 K-blocks, A1: KB1 K-blocks (A-fragment layout). B: packed with
// KBtotal = KB0+KB1 blocks per N-tile. One wave per 16x16 C tile.
// z-dim replicates A0/C (used for the 196-slice feat_proj GEMM).
// ---------------------------------------------------------------------------
__global__ void gemm32_wmma_kernel(const __bf16* __restrict__ A0, int KB0,
                                   const __bf16* __restrict__ A1, int KB1,
                                   const __bf16* __restrict__ Bp,
                                   const float*  __restrict__ bias,
                                   float* __restrict__ C, int ldC, int N,
                                   int64_t strideA0z, int64_t strideCz)
{
    int z = blockIdx.y;
    A0 += (int64_t)z * strideA0z;
    C  += (int64_t)z * strideCz;

    int wave = blockIdx.x * (blockDim.x >> 5) + (threadIdx.x >> 5);
    int lane = threadIdx.x & 31;
    int mt = wave & 1;
    int nb = wave >> 1;
    if (nb * 16 >= N) return;

    int KB = KB0 + KB1;
    const v16bf* Bv = (const v16bf*)(Bp + (int64_t)nb * KB * 512);

    v8f acc = {};
    for (int kb = 0; kb < KB0; ++kb) {
        v16bf a = *(const v16bf*)(A0 + (((int64_t)kb * 2 + mt) * 32 + lane) * 16);
        v16bf b = Bv[(int64_t)kb * 32 + lane];
        acc = __builtin_amdgcn_wmma_f32_16x16x32_bf16(false, a, false, b,
                                                      (short)0, acc, false, false);
    }
    for (int kb = 0; kb < KB1; ++kb) {
        v16bf a = *(const v16bf*)(A1 + (((int64_t)kb * 2 + mt) * 32 + lane) * 16);
        v16bf b = Bv[(int64_t)(KB0 + kb) * 32 + lane];
        acc = __builtin_amdgcn_wmma_f32_16x16x32_bf16(false, a, false, b,
                                                      (short)0, acc, false, false);
    }

    int n  = nb * 16 + (lane & 15);
    float bv = bias ? bias[n] : 0.f;
    int mbase = mt * 16 + (((lane >> 4) & 1) << 3);      // C layout: VGPR r -> row
    #pragma unroll
    for (int r = 0; r < 8; ++r) {
        C[(int64_t)(mbase + r) * ldC + n] = acc[r] + bv;
    }
}

// ---------------------------------------------------------------------------
// Attention scores: score[b,p] = sum_a v[a]*tanh(feat_proj[b,p,a]+hW[b,a]) + bf
// One wave per (b,p).
// ---------------------------------------------------------------------------
__global__ void att_scores_kernel(const float* __restrict__ feat_proj,
                                  const float* __restrict__ hW,
                                  const float* __restrict__ v_att,
                                  const float* __restrict__ b_full,
                                  float* __restrict__ scores)
{
    int wave = blockIdx.x * (blockDim.x >> 5) + (threadIdx.x >> 5);
    int lane = threadIdx.x & 31;
    if (wave >= BB * PP) return;
    int b = wave / PP;
    const float* fp = feat_proj + (int64_t)wave * ATT;
    const float* hw = hW + (int64_t)b * ATT;
    float s = 0.f;
    for (int a = lane; a < ATT; a += 32)
        s += v_att[a] * tanhf(fp[a] + hw[a]);
    #pragma unroll
    for (int off = 16; off > 0; off >>= 1)
        s += __shfl_xor(s, off, 32);
    if (lane == 0) scores[wave] = s + b_full[0];
}

// ---------------------------------------------------------------------------
// Per-batch softmax over P, context = alpha . features, build x=[emb_t|ctx]
// and emit it directly in WMMA A-fragment layout (K=768 -> 24 K-blocks).
// One 256-thread block per b.
// ---------------------------------------------------------------------------
__global__ void softmax_ctx_pack_kernel(const float* __restrict__ scores,
                                        const float* __restrict__ features,
                                        const float* __restrict__ emb,
                                        const long long* __restrict__ captions,
                                        int t, __bf16* __restrict__ packedX)
{
    __shared__ float sc[PP];
    __shared__ float xbuf[EE + ENC];
    __shared__ float mx, sum;
    int b = blockIdx.x;
    int tid = threadIdx.x;

    if (tid < PP) sc[tid] = scores[b * PP + tid];
    __syncthreads();
    if (tid == 0) {
        float m = sc[0];
        for (int p = 1; p < PP; ++p) m = fmaxf(m, sc[p]);
        mx = m;
    }
    __syncthreads();
    if (tid < PP) sc[tid] = __expf(sc[tid] - mx);
    __syncthreads();
    if (tid == 0) {
        float s = 0.f;
        for (int p = 0; p < PP; ++p) s += sc[p];
        sum = 1.0f / s;
    }
    __syncthreads();
    if (tid < PP) sc[tid] *= sum;
    __syncthreads();

    // x[0:256] = emb[captions[b,t]]
    long long tok = captions[(int64_t)b * SS + t];
    if (tid < EE) xbuf[tid] = emb[(int64_t)tok * EE + tid];
    // x[256:768] = context
    for (int e = tid; e < ENC; e += blockDim.x) {
        float acc = 0.f;
        const float* fb = features + ((int64_t)b * PP) * ENC + e;
        for (int p = 0; p < PP; ++p) acc += sc[p] * fb[(int64_t)p * ENC];
        xbuf[EE + e] = acc;
    }
    __syncthreads();

    // scatter row b of x into A-fragment layout
    int mt = b >> 4;
    for (int k = tid; k < EE + ENC; k += blockDim.x) {
        int kb = k >> 5, kin = k & 31;
        int i    = ((kin >> 4) << 3) | (kin & 7);
        int lane = (b & 15) + (((kin >> 3) & 1) << 4);
        packedX[(((int64_t)kb * 2 + mt) * 32 + lane) * 16 + i] = (__bf16)xbuf[k];
    }
}

// ---------------------------------------------------------------------------
// LSTM pointwise: update c (f32), produce h_new directly as packed A-fragments.
// gates layout [32, 2048] = [i | f | g | o].
// ---------------------------------------------------------------------------
__global__ void lstm_pointwise_kernel(const float* __restrict__ gates,
                                      float* __restrict__ c,
                                      __bf16* __restrict__ packedH)
{
    int idx = blockIdx.x * blockDim.x + threadIdx.x;     // 32*512
    if (idx >= BB * DEC) return;
    int b = idx >> 9, d = idx & 511;
    const float* g = gates + (int64_t)b * (4 * DEC);
    float i_ = sigmoidf(g[d]);
    float f_ = sigmoidf(g[d + DEC]);
    float g_ = tanhf(g[d + 2 * DEC]);
    float o_ = sigmoidf(g[d + 3 * DEC]);
    float cn = f_ * c[idx] + i_ * g_;
    float hn = o_ * tanhf(cn);
    c[idx] = cn;
    int kb = d >> 5, kin = d & 31;
    int i    = ((kin >> 4) << 3) | (kin & 7);
    int lane = (b & 15) + (((kin >> 3) & 1) << 4);
    int mt   = b >> 4;
    packedH[(((int64_t)kb * 2 + mt) * 32 + lane) * 16 + i] = (__bf16)hn;
}

// ---------------------------------------------------------------------------
// Host orchestration
// ---------------------------------------------------------------------------
static inline size_t alignup(size_t x) { return (x + 255) & ~(size_t)255; }

extern "C" void kernel_launch(void* const* d_in, const int* in_sizes, int n_in,
                              void* d_out, int out_size, void* d_ws, size_t ws_size,
                              hipStream_t stream)
{
    (void)in_sizes; (void)n_in; (void)out_size; (void)ws_size;
    const float*     features  = (const float*)d_in[0];
    const long long* captions  = (const long long*)d_in[1];
    const float*     emb       = (const float*)d_in[2];
    const float*     W_enc_att = (const float*)d_in[3];
    const float*     b_enc_att = (const float*)d_in[4];
    const float*     W_dec_att = (const float*)d_in[5];
    const float*     b_dec_att = (const float*)d_in[6];
    const float*     v_att     = (const float*)d_in[7];
    const float*     b_full    = (const float*)d_in[8];
    const float*     W_init_h  = (const float*)d_in[9];
    const float*     b_init_h  = (const float*)d_in[10];
    const float*     W_init_c  = (const float*)d_in[11];
    const float*     b_init_c  = (const float*)d_in[12];
    const float*     W_ih      = (const float*)d_in[13];
    const float*     W_hh      = (const float*)d_in[14];
    const float*     b_ih      = (const float*)d_in[15];
    const float*     b_hh      = (const float*)d_in[16];
    const float*     W_fcn     = (const float*)d_in[17];
    const float*     b_fcn     = (const float*)d_in[18];
    float* out = (float*)d_out;

    // ---- workspace carve-up (bf16 packed buffers + f32 intermediates) ----
    char* ws = (char*)d_ws;
    size_t off = 0;
    auto take = [&](size_t bytes) { char* p = ws + off; off += alignup(bytes); return p; };

    __bf16* pW_dec   = (__bf16*)take((size_t)32 * 16 * 512 * 2);          // 512x512
    __bf16* pW_enc   = (__bf16*)take((size_t)32 * 16 * 512 * 2);
    __bf16* pW_ih0   = (__bf16*)take((size_t)32 * 16 * 512 * 2);          // W_init_h
    __bf16* pW_ic0   = (__bf16*)take((size_t)32 * 16 * 512 * 2);          // W_init_c
    __bf16* pW_gates = (__bf16*)take((size_t)128 * 40 * 512 * 2);         // [Wih^T;Whh^T] K=1280,N=2048
    __bf16* pW_fcn   = (__bf16*)take((size_t)625 * 16 * 512 * 2);         // 512x10000
    __bf16* pFeat    = (__bf16*)take((size_t)PP * 16 * 1024 * 2);         // per-p A frags
    float*  featproj = (float*)take((size_t)BB * PP * ATT * 4);
    float*  meanf    = (float*)take((size_t)BB * ENC * 4);
    __bf16* pMean    = (__bf16*)take((size_t)16 * 1024 * 2);
    float*  hbuf     = (float*)take((size_t)BB * DEC * 4);
    float*  cbuf     = (float*)take((size_t)BB * DEC * 4);
    float*  hW       = (float*)take((size_t)BB * ATT * 4);
    float*  scores   = (float*)take((size_t)BB * PP * 4);
    float*  gates    = (float*)take((size_t)BB * 4 * DEC * 4);
    float*  bias_g   = (float*)take((size_t)4 * DEC * 4);
    __bf16* pX       = (__bf16*)take((size_t)24 * 1024 * 2);
    __bf16* pH       = (__bf16*)take((size_t)16 * 1024 * 2);

    const int T256 = 256;
    auto packBlocks = [](int K, int N) { return ((N >> 4) * (K >> 5) + 7) / 8; };

    // ---- one-time prep (re-run every launch; deterministic) ----
    hipLaunchKernelGGL(pack_B_kernel, dim3(packBlocks(512, 512)), dim3(T256), 0, stream,
                       W_dec_att, pW_dec, 512, 512, 512, 0, 0, 16);
    hipLaunchKernelGGL(pack_B_kernel, dim3(packBlocks(512, 512)), dim3(T256), 0, stream,
                       W_enc_att, pW_enc, 512, 512, 512, 0, 0, 16);
    hipLaunchKernelGGL(pack_B_kernel, dim3(packBlocks(512, 512)), dim3(T256), 0, stream,
                       W_init_h, pW_ih0, 512, 512, 512, 0, 0, 16);
    hipLaunchKernelGGL(pack_B_kernel, dim3(packBlocks(512, 512)), dim3(T256), 0, stream,
                       W_init_c, pW_ic0, 512, 512, 512, 0, 0, 16);
    // gates B = [W_ih^T (K=768) ; W_hh^T (K=512)], KBtotal = 40
    hipLaunchKernelGGL(pack_B_kernel, dim3(packBlocks(768, 2048)), dim3(T256), 0, stream,
                       W_ih, pW_gates, 768, 2048, 768, 1, 0, 40);
    hipLaunchKernelGGL(pack_B_kernel, dim3(packBlocks(512, 2048)), dim3(T256), 0, stream,
                       W_hh, pW_gates, 512, 2048, 512, 1, 24, 40);
    hipLaunchKernelGGL(pack_B_kernel, dim3(packBlocks(512, VV)), dim3(T256), 0, stream,
                       W_fcn, pW_fcn, 512, VV, VV, 0, 0, 16);
    hipLaunchKernelGGL(add_bias_kernel, dim3((4 * DEC + 255) / 256), dim3(T256), 0, stream,
                       b_ih, b_hh, bias_g, 4 * DEC);

    // h0/c0 from mean features
    hipLaunchKernelGGL(mean_feat_kernel, dim3((BB * ENC + 255) / 256), dim3(T256), 0, stream,
                       features, meanf);
    hipLaunchKernelGGL(pack_A32_kernel, dim3((BB * ENC + 255) / 256), dim3(T256), 0, stream,
                       meanf, pMean, ENC);
    hipLaunchKernelGGL(gemm32_wmma_kernel, dim3(16, 1), dim3(128), 0, stream,
                       pMean, 16, (const __bf16*)nullptr, 0, pW_ih0, b_init_h,
                       hbuf, DEC, DEC, (int64_t)0, (int64_t)0);
    hipLaunchKernelGGL(gemm32_wmma_kernel, dim3(16, 1), dim3(128), 0, stream,
                       pMean, 16, (const __bf16*)nullptr, 0, pW_ic0, b_init_c,
                       cbuf, DEC, DEC, (int64_t)0, (int64_t)0);
    hipLaunchKernelGGL(pack_A32_kernel, dim3((BB * DEC + 255) / 256), dim3(T256), 0, stream,
                       hbuf, pH, DEC);

    // feat_proj = features @ W_enc_att + b  (196 z-slices of an M=32 GEMM)
    hipLaunchKernelGGL(pack_feat_kernel,
                       dim3((int)(((int64_t)BB * PP * ENC + 255) / 256)), dim3(T256), 0, stream,
                       features, pFeat);
    hipLaunchKernelGGL(gemm32_wmma_kernel, dim3(16, PP), dim3(128), 0, stream,
                       pFeat, 16, (const __bf16*)nullptr, 0, pW_enc, b_enc_att,
                       featproj, PP * ATT, ATT, (int64_t)(16 * 1024), (int64_t)ATT);

    // ---- 127 sequential decode steps ----
    for (int t = 0; t < TSTEPS; ++t) {
        // hW = h @ W_dec_att + b_dec_att
        hipLaunchKernelGGL(gemm32_wmma_kernel, dim3(16, 1), dim3(128), 0, stream,
                           pH, 16, (const __bf16*)nullptr, 0, pW_dec, b_dec_att,
                           hW, ATT, ATT, (int64_t)0, (int64_t)0);
        // additive attention scores
        hipLaunchKernelGGL(att_scores_kernel, dim3((BB * PP) / 8), dim3(T256), 0, stream,
                           featproj, hW, v_att, b_full, scores);
        // softmax + context + embedding -> packed x
        hipLaunchKernelGGL(softmax_ctx_pack_kernel, dim3(BB), dim3(T256), 0, stream,
                           scores, features, emb, captions, t, pX);
        // gates = [x|h] @ [W_ih;W_hh]^T + (b_ih+b_hh)
        hipLaunchKernelGGL(gemm32_wmma_kernel, dim3(64, 1), dim3(128), 0, stream,
                           pX, 24, pH, 16, pW_gates, bias_g,
                           gates, 4 * DEC, 4 * DEC, (int64_t)0, (int64_t)0);
        // LSTM pointwise -> c, packed h
        hipLaunchKernelGGL(lstm_pointwise_kernel, dim3((BB * DEC + 255) / 256), dim3(T256),
                           0, stream, gates, cbuf, pH);
        // logits_t = h @ W_fcn + b_fcn  -> d_out[:, t, :]
        hipLaunchKernelGGL(gemm32_wmma_kernel, dim3(313, 1), dim3(128), 0, stream,
                           pH, 16, (const __bf16*)nullptr, 0, pW_fcn, b_fcn,
                           out + (int64_t)t * VV, TSTEPS * VV, VV, (int64_t)0, (int64_t)0);
    }
}
